// Model_60146722013498
// MI455X (gfx1250) — matrix operand order repaired and checked by reading, not compiled, not run on hardware
//
#include <hip/hip_runtime.h>

typedef __bf16 bf16_t;
typedef __attribute__((ext_vector_type(4)))  __bf16 v4bf;
typedef __attribute__((ext_vector_type(8)))  __bf16 v8bf;
typedef __attribute__((ext_vector_type(16))) __bf16 v16bf;
typedef __attribute__((ext_vector_type(4)))  float  v4f;
typedef __attribute__((ext_vector_type(8)))  float  v8f;

#define D_DIM   128
#define NEXP    4
#define TILE_T  128
#define XROW    136   // padded LDS row stride (bf16 elems): 272B -> bank-conflict-free b128

// ---------------- pre-kernel: f32 weights -> bf16, transposed to [n][e][d] ----------------
__global__ __launch_bounds__(256)
void cvt_weights(const float* __restrict__ W1, const float* __restrict__ W2,
                 bf16_t* __restrict__ Wt) {
  int idx = blockIdx.x * blockDim.x + threadIdx.x;     // over NEXP*D*D, laid out n,e,d
  const int total = NEXP * D_DIM * D_DIM;
  if (idx >= total) return;
  int n = idx >> 14;          // / (128*128)
  int r = idx & 16383;
  int e = r >> 7;
  int d = r & 127;
  Wt[idx]         = (bf16_t)W1[n * 16384 + d * 128 + e];   // W1t[n][e][d]
  Wt[total + idx] = (bf16_t)W2[n * 16384 + d * 128 + e];   // W2t[n][e][d]
}

// ---------------- main fused MoE kernel ----------------
__global__ __launch_bounds__(256)
void moe_kernel(const float* __restrict__ x,
                const bf16_t* __restrict__ Wt,   // [2][NEXP][128][128] bf16, e-major
                const float* __restrict__ b1,
                const float* __restrict__ b2,
                const float* __restrict__ Wg,    // [128][4]
                const float* __restrict__ bg,    // [4]
                float* __restrict__ out) {
  __shared__ __align__(16) unsigned short sX[TILE_T * XROW];     // 34816 B
  __shared__ __align__(16) unsigned short sH[8 * 16 * XROW];     // 34816 B, per-wave slabs
  __shared__ float sG[TILE_T * NEXP];                            // 2048 B

  const int tid  = threadIdx.x;
  const int lane = tid & 31;
  const int wave = tid >> 5;
  const int half = lane >> 4;
  const int l16  = lane & 15;
  const long tileBase = (long)blockIdx.x * TILE_T;

  // ---- load X tile (f32, coalesced b128) -> convert -> bf16 LDS ----
#pragma unroll
  for (int i = 0; i < 16; ++i) {
    int idx = tid + i * 256;            // 4096 groups of 4 floats
    int row = idx >> 5;                 // 32 groups per 128-wide row
    int c4  = (idx & 31) << 2;
    v4f xv = *(const v4f*)(x + (tileBase + row) * D_DIM + c4);
    v4bf bv = { (bf16_t)xv.x, (bf16_t)xv.y, (bf16_t)xv.z, (bf16_t)xv.w };
    *(v4bf*)(&sX[row * XROW + c4]) = bv;
  }
  __syncthreads();

  // ---- gate: logits + softmax over 4 experts (one thread per token) ----
  if (tid < TILE_T) {
    float a0 = bg[0], a1 = bg[1], a2 = bg[2], a3 = bg[3];
    const bf16_t* xr = (const bf16_t*)&sX[tid * XROW];
#pragma unroll 8
    for (int d = 0; d < D_DIM; ++d) {
      float xv = (float)xr[d];
      a0 += xv * Wg[d * NEXP + 0];
      a1 += xv * Wg[d * NEXP + 1];
      a2 += xv * Wg[d * NEXP + 2];
      a3 += xv * Wg[d * NEXP + 3];
    }
    float m  = fmaxf(fmaxf(a0, a1), fmaxf(a2, a3));
    float e0 = __expf(a0 - m), e1 = __expf(a1 - m);
    float e2 = __expf(a2 - m), e3 = __expf(a3 - m);
    float inv = 1.0f / (e0 + e1 + e2 + e3);
    sG[tid * 4 + 0] = e0 * inv;
    sG[tid * 4 + 1] = e1 * inv;
    sG[tid * 4 + 2] = e2 * inv;
    sG[tid * 4 + 3] = e3 * inv;
  }
  __syncthreads();

  const int tokBase = wave * 16;                                   // slab within tile
  const bf16_t* sXrow  = (const bf16_t*)&sX[(tokBase + l16) * XROW];
  bf16_t*       sHbase = (bf16_t*)&sH[wave * 16 * XROW];
  const bf16_t* sHrow  = sHbase + l16 * XROW;

  // A fragments of X (invariant across experts). A layout used here (contiguous k):
  // lanes 0-15: K 0..15; lanes 16-31: K 16..31
  v16bf AX[4];
#pragma unroll
  for (int k = 0; k < 4; ++k) {
    v8bf a0 = *(const v8bf*)(sXrow + k * 32 + half * 16);
    v8bf a1 = *(const v8bf*)(sXrow + k * 32 + 8 + half * 16);
    AX[k] = __builtin_shufflevector(a0, a1, 0,1,2,3,4,5,6,7,8,9,10,11,12,13,14,15);
  }

  v8f outAcc[8];
#pragma unroll
  for (int j = 0; j < 8; ++j) outAcc[j] = (v8f){0.f,0.f,0.f,0.f,0.f,0.f,0.f,0.f};

  for (int n = 0; n < NEXP; ++n) {
    const bf16_t* W1t = Wt + (long)n * D_DIM * D_DIM;
    const bf16_t* W2t = Wt + (long)(NEXP + n) * D_DIM * D_DIM;

    // ---- stage 1: H = relu(X @ W1 + b1), private 16x128 slab -> LDS bf16 ----
#pragma unroll
    for (int j = 0; j < 8; ++j) {
      float bias = b1[n * D_DIM + j * 16 + l16];
      v8f c = { bias, bias, bias, bias, bias, bias, bias, bias };
      const bf16_t* bcol = W1t + (j * 16 + l16) * D_DIM;
#pragma unroll
      for (int k = 0; k < 4; ++k) {
        // B layout: lanes 0-15 hold K 0..15 of chunk, lanes 16-31 hold K 16..31
        v16bf bf = *(const v16bf*)(bcol + k * 32 + half * 16);
        c = __builtin_amdgcn_wmma_f32_16x16x32_bf16(false, AX[k], false, bf,
                                                    (short)0, c, false, false);
      }
      bf16_t* hcol = sHbase + (half * 8) * XROW + j * 16 + l16;
#pragma unroll
      for (int r = 0; r < 8; ++r) {
        float v = c[r];
        hcol[r * XROW] = (bf16_t)(v > 0.f ? v : 0.f);
      }
    }

    // gate weights for this lane's 8 rows (C-layout: M = half*8 + r)
    float g[8];
#pragma unroll
    for (int r = 0; r < 8; ++r)
      g[r] = sG[(tokBase + half * 8 + r) * 4 + n];

    // A fragments of H (wave-private slab; LDS same-wave ops are in order)
    v16bf AH[4];
#pragma unroll
    for (int k = 0; k < 4; ++k) {
      v8bf a0 = *(const v8bf*)(sHrow + k * 32 + half * 16);
      v8bf a1 = *(const v8bf*)(sHrow + k * 32 + 8 + half * 16);
      AH[k] = __builtin_shufflevector(a0, a1, 0,1,2,3,4,5,6,7,8,9,10,11,12,13,14,15);
    }

    // ---- stage 2: O = H @ W2 + b2 ; out += gate * O ----
#pragma unroll
    for (int j = 0; j < 8; ++j) {
      float bias = b2[n * D_DIM + j * 16 + l16];
      v8f c = { bias, bias, bias, bias, bias, bias, bias, bias };
      const bf16_t* bcol = W2t + (j * 16 + l16) * D_DIM;
#pragma unroll
      for (int k = 0; k < 4; ++k) {
        v16bf bf = *(const v16bf*)(bcol + k * 32 + half * 16);
        c = __builtin_amdgcn_wmma_f32_16x16x32_bf16(false, AH[k], false, bf,
                                                    (short)0, c, false, false);
      }
#pragma unroll
      for (int r = 0; r < 8; ++r)
        outAcc[j][r] += g[r] * c[r];
    }
  }

  // ---- write out (f32): per (r,j), half-wave-contiguous 64B segments ----
#pragma unroll
  for (int r = 0; r < 8; ++r) {
    long gtok = tileBase + tokBase + half * 8 + r;
    float* orow = out + gtok * D_DIM + l16;
#pragma unroll
    for (int j = 0; j < 8; ++j)
      orow[j * 16] = outAcc[j][r];
  }
}

extern "C" void kernel_launch(void* const* d_in, const int* in_sizes, int n_in,
                              void* d_out, int out_size, void* d_ws, size_t ws_size,
                              hipStream_t stream) {
  const float* x  = (const float*)d_in[0];
  const float* W1 = (const float*)d_in[1];
  const float* b1 = (const float*)d_in[2];
  const float* W2 = (const float*)d_in[3];
  const float* b2 = (const float*)d_in[4];
  const float* Wg = (const float*)d_in[5];
  const float* bg = (const float*)d_in[6];
  float* out  = (float*)d_out;
  bf16_t* Wt  = (bf16_t*)d_ws;   // needs 2*4*128*128*2 = 256 KB

  const int totalW = NEXP * D_DIM * D_DIM;
  cvt_weights<<<(totalW + 255) / 256, 256, 0, stream>>>(W1, W2, Wt);

  const long tokens = (long)out_size / D_DIM;          // 131072
  const int  blocks = (int)(tokens / TILE_T);          // 1024
  moe_kernel<<<blocks, 256, 0, stream>>>(x, Wt, b1, b2, Wg, bg, out);
}
